// Poolformer_6640019439815
// MI455X (gfx1250) — matrix operand order repaired and checked by reading, not compiled, not run on hardware
//
#include <hip/hip_runtime.h>
#include <float.h>
#include <math.h>

// ---------------- problem constants ----------------
#define N_TOK 2048
#define DMODEL 512
#define FFDIM 2048
#define VOCAB 32000
#define NW 16      // windows
#define NH 8       // heads
#define WSZ 128    // window size
#define DH 64      // dim per head

typedef _Float16 v16h __attribute__((ext_vector_type(16)));
typedef float    v8f  __attribute__((ext_vector_type(8)));

__device__ __forceinline__ v8f zero8() {
  v8f z;
#pragma unroll
  for (int i = 0; i < 8; ++i) z[i] = 0.f;
  return z;
}

__device__ __forceinline__ v8f wmma32(v16h a, v16h b, v8f c) {
  return __builtin_amdgcn_wmma_f32_16x16x32_f16(false, a, false, b, (short)0, c,
                                                false, false);
}

// A-fragment (16x32 f16) from row-major LDS tile (contiguous k order):
// lanes 0-15: rows M=0..15, K=0..15; lanes 16-31: K=16..31.
// One contiguous 32B chunk per lane -> 2x ds_load_b128.
__device__ __forceinline__ v16h ld_a(const _Float16* p, int stride, int lane) {
  v16h f;
  const _Float16* r = p + (lane & 15) * stride + (lane & 16);
#pragma unroll
  for (int i = 0; i < 16; ++i) f[i] = r[i];
  return f;
}

// B-fragment (32x16 f16) from a TRANSPOSED (N-major, K-contiguous) LDS tile:
// lane holds column n = lane&15, K = (lane&16)+0..15 -> 32 contiguous bytes.
__device__ __forceinline__ v16h ld_bt(const _Float16* p, int stride, int lane) {
  v16h f;
  const _Float16* r = p + (lane & 15) * stride + (lane & 16);
#pragma unroll
  for (int i = 0; i < 16; ++i) f[i] = r[i];
  return f;
}

// ---------------- generic WMMA GEMM: C = act(A @ W + bias) (+R) -------------
// CONTRACT: K % 32 == 0, N % 128 == 0 (true at every call site).
// A: [M,K] stride K (f32). W: [K,N-slice] row stride ldw (f32).
// C,R: [M,N] stride N (f32).
#define BM 128
#define BN 128
#define BK 32
#define AST 40   // f16 stride (80B = 5*16B)
#define BTS 40

template <int ACT, bool RES>
__global__ __launch_bounds__(256) void gemm_kernel(
    const float* __restrict__ A, const float* __restrict__ W,
    const float* __restrict__ bias, const float* __restrict__ R,
    float* __restrict__ C, int M, int N, int K, int ldw) {
  __shared__ _Float16 sA[BM][AST];   // row-major M x K
  __shared__ _Float16 sBT[BN][BTS];  // transposed: N-major, K contiguous
  int tid = threadIdx.x, lane = tid & 31, wv = tid >> 5;
  int m0 = blockIdx.y * BM, n0 = blockIdx.x * BN;

  v8f acc[8];
#pragma unroll
  for (int t = 0; t < 8; ++t) acc[t] = zero8();

  for (int k0 = 0; k0 < K; k0 += BK) {
#pragma unroll
    for (int i = 0; i < 4; ++i) {  // A tile 128x32 via float4
      int s = tid + i * 256;       // 1024 float4 slots
      int r = s >> 3, c4 = s & 7;
      int gm = m0 + r;
      float4 v = make_float4(0.f, 0.f, 0.f, 0.f);
      if (gm < M) v = *(const float4*)(A + (size_t)gm * K + k0 + c4 * 4);
      _Float16* d = &sA[r][c4 * 4];
      d[0] = (_Float16)v.x; d[1] = (_Float16)v.y;
      d[2] = (_Float16)v.z; d[3] = (_Float16)v.w;
    }
#pragma unroll
    for (int i = 0; i < 4; ++i) {  // B tile 32x128 -> transposed store
      int s = tid + i * 256;       // (k4, n): n fastest for coalescing
      int n = s & 127, k4 = s >> 7;
      int gn = n0 + n;
      _Float16* d = &sBT[n][k4 * 4];
#pragma unroll
      for (int kk = 0; kk < 4; ++kk)
        d[kk] = (_Float16)W[(size_t)(k0 + k4 * 4 + kk) * ldw + gn];
    }
    __syncthreads();
    v16h af = ld_a(&sA[wv * 16][0], AST, lane);
#pragma unroll
    for (int t = 0; t < 8; ++t) {
      v16h bf = ld_bt(&sBT[t * 16][0], BTS, lane);
      acc[t] = wmma32(af, bf, acc[t]);
    }
    __syncthreads();
  }

  int mo = wv * 16 + ((lane >> 4) << 3);
  int nc = lane & 15;
#pragma unroll
  for (int t = 0; t < 8; ++t) {
    int gn = n0 + t * 16 + nc;
    float bv = bias ? bias[gn] : 0.f;
#pragma unroll
    for (int v = 0; v < 8; ++v) {
      int gm = m0 + mo + v;
      if (gm >= M) continue;
      float c = acc[t][v] + bv;
      if (ACT == 1) c = 0.5f * c * (1.f + erff(c * 0.70710678118654752f));
      if (RES) c += R[(size_t)gm * N + gn];
      C[(size_t)gm * N + gn] = c;
    }
  }
}

static void gemm(hipStream_t st, const float* A, const float* W,
                 const float* bias, const float* R, float* C, int M, int N,
                 int K, int ldw, int act) {
  dim3 g((N + BN - 1) / BN, (M + BM - 1) / BM);
  if (R) {
    if (act) gemm_kernel<1, true><<<g, 256, 0, st>>>(A, W, bias, R, C, M, N, K, ldw);
    else     gemm_kernel<0, true><<<g, 256, 0, st>>>(A, W, bias, R, C, M, N, K, ldw);
  } else {
    if (act) gemm_kernel<1, false><<<g, 256, 0, st>>>(A, W, bias, R, C, M, N, K, ldw);
    else     gemm_kernel<0, false><<<g, 256, 0, st>>>(A, W, bias, R, C, M, N, K, ldw);
  }
}

// ---------------- small elementwise / reduction kernels ----------------
__global__ __launch_bounds__(256) void embed_kernel(
    const int* __restrict__ tokens, const float* __restrict__ emb,
    float* __restrict__ x) {
  int i = blockIdx.x * 256 + threadIdx.x;
  if (i < N_TOK * DMODEL) {
    int r = i >> 9, c = i & 511;
    x[i] = emb[(size_t)tokens[r] * DMODEL + c];
  }
}

__global__ __launch_bounds__(256) void pool_kernel(
    const float* __restrict__ x, const float* __restrict__ gpos,
    float* __restrict__ g) {
  int w = blockIdx.x;
  for (int c = threadIdx.x; c < DMODEL; c += 256) {
    float s = 0.f;
    for (int i = 0; i < WSZ; ++i) s += x[(size_t)(w * WSZ + i) * DMODEL + c];
    g[(size_t)w * DMODEL + c] = s * (1.f / 128.f) + gpos[(size_t)w * DMODEL + c];
  }
}

__global__ __launch_bounds__(256) void ln_kernel(
    const float* __restrict__ in, const float* __restrict__ w,
    const float* __restrict__ b, float* __restrict__ out) {
  __shared__ float red[256];
  int r = blockIdx.x, tid = threadIdx.x;
  const float* xr = in + (size_t)r * DMODEL;
  float a = xr[tid], c = xr[tid + 256];
  red[tid] = a + c;
  __syncthreads();
  for (int s = 128; s > 0; s >>= 1) {
    if (tid < s) red[tid] += red[tid + s];
    __syncthreads();
  }
  float mu = red[0] * (1.f / 512.f);
  __syncthreads();
  float da = a - mu, dc = c - mu;
  red[tid] = da * da + dc * dc;
  __syncthreads();
  for (int s = 128; s > 0; s >>= 1) {
    if (tid < s) red[tid] += red[tid + s];
    __syncthreads();
  }
  float inv = rsqrtf(red[0] * (1.f / 512.f) + 1e-5f);
  out[(size_t)r * DMODEL + tid]       = da * inv * w[tid] + b[tid];
  out[(size_t)r * DMODEL + tid + 256] = dc * inv * w[tid + 256] + b[tid + 256];
}

__global__ __launch_bounds__(256) void ctx_build_kernel(
    const float* __restrict__ gn, const float* __restrict__ x,
    float* __restrict__ ctx) {
  int i = blockIdx.x * 256 + threadIdx.x;
  if (i >= NW * 129 * DMODEL) return;
  int r = i / DMODEL, c = i % DMODEL;
  int w = r / 129, rr = r % 129;
  ctx[i] = (rr == 0) ? gn[(size_t)w * DMODEL + c]
                     : x[(size_t)(w * WSZ + rr - 1) * DMODEL + c];
}

// rotary: pairs (d, d+32) per head; freq index = d (0..31)
__global__ __launch_bounds__(256) void rotary_kernel(float* __restrict__ q,
                                                     float* __restrict__ k) {
  int idx = blockIdx.x * 256 + threadIdx.x;  // N_TOK * NH * 32
  if (idx >= N_TOK * NH * 32) return;
  int pos = idx >> 8;
  int rem = idx & 255;
  int h = rem >> 5, j = rem & 31;
  // 10000^(-2j/64) = exp2(-j * (2/64) * log2(10000))
  float invf = exp2f(-(float)j * (2.f / 64.f) * 13.287712379549449f);
  float a = (float)pos * invf;
  float c = cosf(a), s = sinf(a);
  size_t base = (size_t)pos * DMODEL + h * DH + j;
  float q1 = q[base], q2 = q[base + 32];
  q[base] = q1 * c - q2 * s;
  q[base + 32] = q2 * c + q1 * s;
  float k1 = k[base], k2 = k[base + 32];
  k[base] = k1 * c - k2 * s;
  k[base + 32] = k2 * c + k1 * s;
}

// global-token attention: 1 query per (window, head) over 129 ctx keys
__global__ __launch_bounds__(256) void gattn_kernel(
    const float* __restrict__ gq, const float* __restrict__ ckv,
    float* __restrict__ gat) {
  __shared__ float sc[129];
  __shared__ float sq[DH];
  __shared__ float sm, ss;
  int w = blockIdx.x, h = blockIdx.y, tid = threadIdx.x;
  if (tid < DH) sq[tid] = gq[(size_t)w * DMODEL + h * DH + tid];
  __syncthreads();
  if (tid < 129) {
    const float* kr = ckv + (size_t)(w * 129 + tid) * (2 * DMODEL) + h * DH;
    float s = 0.f;
    for (int d = 0; d < DH; ++d) s += sq[d] * kr[d];
    sc[tid] = s * 0.125f;
  }
  __syncthreads();
  if (tid == 0) {
    float m = -FLT_MAX;
    for (int j = 0; j < 129; ++j) m = fmaxf(m, sc[j]);
    float sum = 0.f;
    for (int j = 0; j < 129; ++j) sum += expf(sc[j] - m);
    sm = m; ss = sum;
  }
  __syncthreads();
  if (tid < DH) {
    float inv = 1.f / ss;
    float o = 0.f;
    for (int j = 0; j < 129; ++j) {
      float p = expf(sc[j] - sm) * inv;
      o += p * ckv[(size_t)(w * 129 + j) * (2 * DMODEL) + DMODEL + h * DH + tid];
    }
    gat[(size_t)w * DMODEL + h * DH + tid] = o;
  }
}

// ---------------- flash-style local windowed attention (WMMA) ---------------
// One block per (window w, head h). 8 waves, 16 rows each.
// Unified key array s_kf rows: j<15 -> global tokens gk, j==15 -> pad (always
// masked), j in [16,271] -> kcat (prev|cur window keys; values == keys per the
// reference's vcat=kcat bug), j in [272,287] -> zero pad (masked by causal
// mask automatically). s_vT holds vf transposed (d-major) so PV-phase WMMA
// B-fragments are contiguous ds_load_b128s.
// Mask (branchless): masked = ((j<15) & (j>=w)) | (j==15) | (j > iw+144)
//   - local term: (j-16) > iw+128  <=>  j > iw+144; auto-false for j<16.
#define KST 72    // 144B = 9*16B
#define VTS 296   // 592B = 37*16B
#define PST 40

__global__ __launch_bounds__(256) void local_attn_kernel(
    const float* __restrict__ q, const float* __restrict__ k,
    const float* __restrict__ gkv, float* __restrict__ out) {
  __shared__ _Float16 s_kf[288][KST];  // keys, row-major (d contiguous)
  __shared__ _Float16 s_vT[DH][VTS];   // values transposed (j contiguous)
  __shared__ _Float16 s_p[8][16][PST]; // per-wave P chunk (16x32)

  int w = blockIdx.x, h = blockIdx.y;
  int tid = threadIdx.x, lane = tid & 31, wv = tid >> 5;
  int m0 = wv * 16;

  // stage keys + transposed values
  for (int e = tid; e < 288 * 16; e += 256) {
    int r = e >> 4, c4 = e & 15;  // row, float4 column
    float4 kv = make_float4(0.f, 0.f, 0.f, 0.f);
    float4 vv = make_float4(0.f, 0.f, 0.f, 0.f);
    if (r < 15) {
      kv = *(const float4*)(gkv + (size_t)r * (2 * DMODEL) + h * DH + c4 * 4);
      vv = *(const float4*)(gkv + (size_t)r * (2 * DMODEL) + DMODEL + h * DH + c4 * 4);
    } else if (r >= 16 && r < 272) {
      int rr = r - 16;
      int srcw = (rr < WSZ) ? (w - 1) : w;
      if (srcw >= 0) {
        kv = *(const float4*)(k + (size_t)(srcw * WSZ + (rr & 127)) * DMODEL +
                              h * DH + c4 * 4);
        vv = kv;  // faithful to reference: windowed values are the keys
      }
    }
    _Float16* kd = &s_kf[r][c4 * 4];
    kd[0] = (_Float16)kv.x; kd[1] = (_Float16)kv.y;
    kd[2] = (_Float16)kv.z; kd[3] = (_Float16)kv.w;
    int db = c4 * 4;
    s_vT[db + 0][r] = (_Float16)vv.x;
    s_vT[db + 1][r] = (_Float16)vv.y;
    s_vT[db + 2][r] = (_Float16)vv.z;
    s_vT[db + 3][r] = (_Float16)vv.w;
  }
  __syncthreads();

  // q fragments straight from global (A layout, K=0..31 / 32..63)
  int qrow = w * WSZ + m0 + (lane & 15);
  int kb8 = (lane & 16) >> 1;
  const float* qp = q + (size_t)qrow * DMODEL + h * DH;
  float4 q0 = *(const float4*)(qp + kb8);
  float4 q1 = *(const float4*)(qp + kb8 + 4);
  float4 q2 = *(const float4*)(qp + 16 + kb8);
  float4 q3 = *(const float4*)(qp + 16 + kb8 + 4);
  float4 q4 = *(const float4*)(qp + 32 + kb8);
  float4 q5 = *(const float4*)(qp + 32 + kb8 + 4);
  float4 q6 = *(const float4*)(qp + 48 + kb8);
  float4 q7 = *(const float4*)(qp + 48 + kb8 + 4);
  v16h a0, a1;
  a0[0]  = (_Float16)q0.x; a0[1]  = (_Float16)q0.y; a0[2]  = (_Float16)q0.z; a0[3]  = (_Float16)q0.w;
  a0[4]  = (_Float16)q1.x; a0[5]  = (_Float16)q1.y; a0[6]  = (_Float16)q1.z; a0[7]  = (_Float16)q1.w;
  a0[8]  = (_Float16)q2.x; a0[9]  = (_Float16)q2.y; a0[10] = (_Float16)q2.z; a0[11] = (_Float16)q2.w;
  a0[12] = (_Float16)q3.x; a0[13] = (_Float16)q3.y; a0[14] = (_Float16)q3.z; a0[15] = (_Float16)q3.w;
  a1[0]  = (_Float16)q4.x; a1[1]  = (_Float16)q4.y; a1[2]  = (_Float16)q4.z; a1[3]  = (_Float16)q4.w;
  a1[4]  = (_Float16)q5.x; a1[5]  = (_Float16)q5.y; a1[6]  = (_Float16)q5.z; a1[7]  = (_Float16)q5.w;
  a1[8]  = (_Float16)q6.x; a1[9]  = (_Float16)q6.y; a1[10] = (_Float16)q6.z; a1[11] = (_Float16)q6.w;
  a1[12] = (_Float16)q7.x; a1[13] = (_Float16)q7.y; a1[14] = (_Float16)q7.z; a1[15] = (_Float16)q7.w;

  int half = (lane >> 4) << 3;  // 0 or 8 (C-layout row offset)
  int nc = lane & 15;
  int kb16 = lane & 16;

  float m_run[8], l_run[8];
#pragma unroll
  for (int v = 0; v < 8; ++v) { m_run[v] = -FLT_MAX; l_run[v] = 0.f; }
  v8f oacc[4];
#pragma unroll
  for (int t = 0; t < 4; ++t) oacc[t] = zero8();

  for (int s = 0; s < 9; ++s) {  // 9 chunks of 32 keys (288 padded)
    int jb = s * 32;
    int j0 = jb + nc;       // lane's column in N-tile 0
    int j1 = jb + 16 + nc;  // lane's column in N-tile 1
    // j-only mask term (global tokens + pad column), branchless
    int mgp0 = (int)((j0 < 15) & (j0 >= w)) | (int)(j0 == 15);
    int mgp1 = (int)((j1 < 15) & (j1 >= w)) | (int)(j1 == 15);

    v8f sa[2];
    sa[0] = zero8(); sa[1] = zero8();
#pragma unroll
    for (int nt = 0; nt < 2; ++nt) {  // sim = q @ kf^T
      int j = jb + nt * 16 + nc;
      const _Float16* r0 = &s_kf[j][kb8];
      const _Float16* r1 = &s_kf[j][32 + kb8];
      v16h b0, b1;
#pragma unroll
      for (int i = 0; i < 16; ++i) { b0[i] = r0[i + (i & 8)]; b1[i] = r1[i + (i & 8)]; }
      sa[nt] = wmma32(a0, b0, sa[nt]);
      sa[nt] = wmma32(a1, b1, sa[nt]);
    }

    float alpha[8];
#pragma unroll
    for (int v = 0; v < 8; ++v) {  // masked online softmax per row
      int thr = m0 + half + v + 144;  // local mask: j > iw + 144
      int msk0 = mgp0 | (int)(j0 > thr);
      int msk1 = mgp1 | (int)(j1 > thr);
      float s0 = msk0 ? -FLT_MAX : sa[0][v] * 0.125f;
      float s1 = msk1 ? -FLT_MAX : sa[1][v] * 0.125f;
      sa[0][v] = s0;
      sa[1][v] = s1;
      float cm = fmaxf(s0, s1);
      for (int d = 1; d < 16; d <<= 1) cm = fmaxf(cm, __shfl_xor(cm, d, 32));
      float mn = fmaxf(m_run[v], cm);
      float al = expf(m_run[v] - mn);
      float p0 = expf(s0 - mn);
      float p1 = expf(s1 - mn);
      sa[0][v] = p0;
      sa[1][v] = p1;
      float csum = p0 + p1;
      for (int d = 1; d < 16; d <<= 1) csum += __shfl_xor(csum, d, 32);
      l_run[v] = l_run[v] * al + csum;
      m_run[v] = mn;
      alpha[v] = al;
      s_p[wv][half + v][nc]      = (_Float16)p0;
      s_p[wv][half + v][16 + nc] = (_Float16)p1;
    }

#pragma unroll
    for (int nt = 0; nt < 4; ++nt)  // rescale running output
#pragma unroll
      for (int v = 0; v < 8; ++v) oacc[nt][v] *= alpha[v];

    // oacc += P_chunk @ vf_chunk  (same-wave LDS ops are in-order: DScnt)
    v16h pf = ld_a(&s_p[wv][0][0], PST, lane);
#pragma unroll
    for (int nt = 0; nt < 4; ++nt) {
      const _Float16* vr = &s_vT[nt * 16 + nc][jb + kb16];
      v16h bv;
#pragma unroll
      for (int i = 0; i < 16; ++i) bv[i] = vr[i];
      oacc[nt] = wmma32(pf, bv, oacc[nt]);
    }
  }

#pragma unroll
  for (int nt = 0; nt < 4; ++nt)
#pragma unroll
    for (int v = 0; v < 8; ++v) {
      int iw = m0 + half + v;
      out[(size_t)(w * WSZ + iw) * DMODEL + h * DH + nt * 16 + nc] =
          oacc[nt][v] / l_run[v];
    }
}

// ---------------- host orchestration ----------------
extern "C" void kernel_launch(void* const* d_in, const int* in_sizes, int n_in,
                              void* d_out, int out_size, void* d_ws,
                              size_t ws_size, hipStream_t stream) {
  (void)in_sizes; (void)n_in; (void)out_size; (void)ws_size;
  const int*   tokens     = (const int*)d_in[0];
  const float* tok_emb    = (const float*)d_in[1];
  const float* gpos_emb   = (const float*)d_in[2];
  const float* g_norm_w   = (const float*)d_in[3];
  const float* g_norm_b   = (const float*)d_in[4];
  const float* g_Wq       = (const float*)d_in[5];
  const float* g_Wkv      = (const float*)d_in[6];
  const float* g_Wo       = (const float*)d_in[7];
  const float* g_bo       = (const float*)d_in[8];
  const float* gff_norm_w = (const float*)d_in[9];
  const float* gff_norm_b = (const float*)d_in[10];
  const float* gff_W1     = (const float*)d_in[11];
  const float* gff_b1     = (const float*)d_in[12];
  const float* gff_W2     = (const float*)d_in[13];
  const float* gff_b2     = (const float*)d_in[14];
  const float* la_norm_w  = (const float*)d_in[15];
  const float* la_norm_b  = (const float*)d_in[16];
  const float* la_Wq      = (const float*)d_in[17];
  const float* la_Wkv     = (const float*)d_in[18];
  const float* la_Wo      = (const float*)d_in[19];
  const float* la_bo      = (const float*)d_in[20];
  const float* lff_norm_w = (const float*)d_in[21];
  const float* lff_norm_b = (const float*)d_in[22];
  const float* lff_W1     = (const float*)d_in[23];
  const float* lff_b1     = (const float*)d_in[24];
  const float* lff_W2     = (const float*)d_in[25];
  const float* lff_b2     = (const float*)d_in[26];
  const float* out_norm_w = (const float*)d_in[27];
  const float* out_norm_b = (const float*)d_in[28];
  const float* out_W      = (const float*)d_in[29];
  const float* out_b      = (const float*)d_in[30];
  float* outp = (float*)d_out;

  float* p = (float*)d_ws;
  float* x    = p; p += (size_t)N_TOK * DMODEL;
  float* xn   = p; p += (size_t)N_TOK * DMODEL;
  float* qb   = p; p += (size_t)N_TOK * DMODEL;
  float* kbuf = p; p += (size_t)N_TOK * DMODEL;
  float* ao   = p; p += (size_t)N_TOK * DMODEL;
  float* ffh  = p; p += (size_t)N_TOK * FFDIM;
  float* g    = p; p += (size_t)NW * DMODEL;
  float* gn   = p; p += (size_t)NW * DMODEL;
  float* gq   = p; p += (size_t)NW * DMODEL;
  float* gat  = p; p += (size_t)NW * DMODEL;
  float* gffh = p; p += (size_t)NW * FFDIM;
  float* ctx  = p; p += (size_t)NW * 129 * DMODEL;
  float* ckv  = p; p += (size_t)NW * 129 * 2 * DMODEL;
  float* gkv  = p; p += (size_t)NW * 2 * DMODEL;

  embed_kernel<<<(N_TOK * DMODEL + 255) / 256, 256, 0, stream>>>(tokens, tok_emb, x);
  pool_kernel<<<NW, 256, 0, stream>>>(x, gpos_emb, g);

  for (int l = 0; l < 2; ++l) {
    // --- global-token transformer (shared weights) ---
    ln_kernel<<<NW, 256, 0, stream>>>(g, g_norm_w, g_norm_b, gn);
    gemm(stream, gn, g_Wq, nullptr, nullptr, gq, NW, DMODEL, DMODEL, DMODEL, 0);
    ctx_build_kernel<<<(NW * 129 * DMODEL + 255) / 256, 256, 0, stream>>>(gn, x, ctx);
    gemm(stream, ctx, g_Wkv, nullptr, nullptr, ckv, NW * 129, 2 * DMODEL, DMODEL,
         2 * DMODEL, 0);
    gattn_kernel<<<dim3(NW, NH), 256, 0, stream>>>(gq, ckv, gat);
    gemm(stream, gat, g_Wo, g_bo, g, g, NW, DMODEL, DMODEL, DMODEL, 0);
    ln_kernel<<<NW, 256, 0, stream>>>(g, gff_norm_w, gff_norm_b, gn);
    gemm(stream, gn, gff_W1, gff_b1, nullptr, gffh, NW, FFDIM, DMODEL, FFDIM, 1);
    gemm(stream, gffh, gff_W2, gff_b2, g, g, NW, DMODEL, FFDIM, DMODEL, 0);

    // --- local windowed attention ---
    ln_kernel<<<N_TOK, 256, 0, stream>>>(x, la_norm_w + (size_t)l * DMODEL,
                                         la_norm_b + (size_t)l * DMODEL, xn);
    gemm(stream, xn, la_Wq + (size_t)l * DMODEL * DMODEL, nullptr, nullptr, qb,
         N_TOK, DMODEL, DMODEL, DMODEL, 0);
    gemm(stream, xn, la_Wkv + (size_t)l * DMODEL * 2 * DMODEL, nullptr, nullptr,
         kbuf, N_TOK, DMODEL, DMODEL, 2 * DMODEL, 0);  // k = first half only
    gemm(stream, g, la_Wkv + (size_t)l * DMODEL * 2 * DMODEL, nullptr, nullptr,
         gkv, NW - 1, 2 * DMODEL, DMODEL, 2 * DMODEL, 0);  // un-normed g[:, :-1]
    rotary_kernel<<<(N_TOK * NH * 32 + 255) / 256, 256, 0, stream>>>(qb, kbuf);
    local_attn_kernel<<<dim3(NW, NH), 256, 0, stream>>>(qb, kbuf, gkv, ao);
    gemm(stream, ao, la_Wo + (size_t)l * DMODEL * DMODEL,
         la_bo + (size_t)l * DMODEL, x, x, N_TOK, DMODEL, DMODEL, DMODEL, 0);
    ln_kernel<<<N_TOK, 256, 0, stream>>>(x, lff_norm_w + (size_t)l * DMODEL,
                                         lff_norm_b + (size_t)l * DMODEL, xn);
    gemm(stream, xn, lff_W1 + (size_t)l * DMODEL * FFDIM,
         lff_b1 + (size_t)l * FFDIM, nullptr, ffh, N_TOK, FFDIM, DMODEL, FFDIM, 1);
    gemm(stream, ffh, lff_W2 + (size_t)l * FFDIM * DMODEL,
         lff_b2 + (size_t)l * DMODEL, x, x, N_TOK, DMODEL, FFDIM, DMODEL, 0);
  }

  // --- LM head (memory-bound: 65.5MB weights + 262MB logits) ---
  ln_kernel<<<N_TOK, 256, 0, stream>>>(x, out_norm_w, out_norm_b, xn);
  gemm(stream, xn, out_W, out_b, nullptr, outp, N_TOK, VOCAB, DMODEL, VOCAB, 0);
}
